// CircuitRankNet_14886356648664
// MI455X (gfx1250) — matrix operand order repaired and checked by reading, not compiled, not run on hardware
//
#include <hip/hip_runtime.h>
#include <math.h>

#define DHID 128
#define GSEG 64

typedef __attribute__((ext_vector_type(16))) __bf16 v16bf;
typedef __attribute__((ext_vector_type(8)))  float  v8f;

struct alignas(16) U4 { unsigned x, y, z, w; };
union Frag { v16bf v; U4 q[2]; };

__device__ __forceinline__ unsigned pk_bf16(float a, float b) {
    unsigned short ua = __builtin_bit_cast(unsigned short, (__bf16)a);
    unsigned short ub = __builtin_bit_cast(unsigned short, (__bf16)b);
    return (unsigned)ua | ((unsigned)ub << 16);
}

// ---------------- utility kernels ----------------
__global__ void fill_f32(float* __restrict__ p, int n, float val) {
    int i = blockIdx.x * blockDim.x + threadIdx.x;
    if (i < n) p[i] = val;
}

__global__ void deg_scatter(const int* __restrict__ dst, float* __restrict__ deg, int E) {
    int i = blockIdx.x * blockDim.x + threadIdx.x;
    if (i < E) atomicAdd(deg + dst[i], 1.0f);
}

__global__ void rsqrt_inplace(float* __restrict__ p, int n) {
    int i = blockIdx.x * blockDim.x + threadIdx.x;
    if (i < n) {
        float d = p[i];
        p[i] = (d > 0.0f) ? rsqrtf(d) : 0.0f;
    }
}

// Pre-swizzle 128x128 f32 weight into per-lane bf16 B-fragment layout.
// Consumption index: i = ((c*4+s)*32 + lane)*16 + e, where
//   k = s*32 + (e>>3)*16 + (lane>>4)*8 + (e&7), n = c*16 + (lane&15)
__global__ void pack_w_bf16(const float* __restrict__ W, unsigned short* __restrict__ out) {
    int i = blockIdx.x * blockDim.x + threadIdx.x;   // 0..16383
    int e    = i & 15;
    int lane = (i >> 4) & 31;
    int t    = i >> 9;            // tile = c*4 + s
    int s    = t & 3;
    int c    = t >> 2;
    int k = s * 32 + ((e >> 3) << 4) + ((lane >> 4) << 3) + (e & 7);
    int n = (c << 4) + (lane & 15);
    out[i] = __builtin_bit_cast(unsigned short, (__bf16)W[k * DHID + n]);
}

// ---------------- WMMA GEMM: C[N x 128] = A[N x 128] * W[128 x 128] ----------------
__global__ __launch_bounds__(256) void gemm_nk128_wmma(
    const float* __restrict__ A, const U4* __restrict__ Wq,
    float* __restrict__ C) {
    __shared__ U4 As[256];        // 16 rows x 128 K of bf16, packed as 16B quads
    const int row0 = blockIdx.x * 16;
    const int tid  = threadIdx.x;

    // Stage A tile -> LDS as bf16. Thread t: row m = t>>4, k base = (t&15)*8.
    {
        int m  = tid >> 4;
        int kb = (tid & 15) * 8;
        const float4* src = (const float4*)(A + (size_t)(row0 + m) * DHID + kb);
        float4 f0 = src[0];
        float4 f1 = src[1];
        U4 q;
        q.x = pk_bf16(f0.x, f0.y);
        q.y = pk_bf16(f0.z, f0.w);
        q.z = pk_bf16(f1.x, f1.y);
        q.w = pk_bf16(f1.z, f1.w);
        As[tid] = q;               // element offset = m*128 + kb  (U4 index == tid)
    }
    __syncthreads();

    const int w    = tid >> 5;     // wave id = column tile (0..7)
    const int lane = tid & 31;
    const int half = lane >> 4;
    const int m    = lane & 15;

    v8f acc = {};
#pragma unroll
    for (int s = 0; s < 4; ++s) {
        Frag fa, fb;
        // A fragment: elements 0..7  -> K = s*32 + half*8 + 0..7
        //             elements 8..15 -> K = s*32 + 16 + half*8 + 0..7
        int abase = (m * DHID + s * 32 + half * 8) >> 3;   // U4 index
        fa.q[0] = As[abase];
        fa.q[1] = As[abase + 2];                           // +16 bf16 elements
        // B fragment: lane-contiguous prepacked weights
        int bbase = ((w * 4 + s) * 32 + lane) * 2;         // U4 index
        fb.q[0] = Wq[bbase];
        fb.q[1] = Wq[bbase + 1];
        acc = __builtin_amdgcn_wmma_f32_16x16x32_bf16(
            false, fa.v, false, fb.v, (short)0, acc, false, false);
    }

    // D layout: VGPR v -> M = v + 8*half, N = lane&15
    float* out = C + (size_t)row0 * DHID + w * 16 + (lane & 15);
#pragma unroll
    for (int v = 0; v < 8; ++v) {
        out[(size_t)(v + 8 * half) * DHID] = acc[v];
    }
}

// ---------------- GCN aggregation ----------------
// Self-loop term + bias: out[i][c] = h[i][c]*dinv[i]^2 + bias[c]
__global__ void self_bias(const float* __restrict__ h, const float* __restrict__ dinv,
                          const float* __restrict__ bias, float* __restrict__ out, int total) {
    int i = blockIdx.x * blockDim.x + threadIdx.x;
    if (i < total) {
        int node = i >> 7;
        int c    = i & 127;
        float di = dinv[node];
        out[i] = h[i] * di * di + bias[c];
    }
}

// Edge scatter: out[dst] += h[src] * dinv[src]*dinv[dst]; 32 lanes per edge, float4 each.
__global__ void agg_edges(const float* __restrict__ h, const int* __restrict__ src,
                          const int* __restrict__ dst, const float* __restrict__ dinv,
                          float* __restrict__ out, int E) {
    unsigned idx = blockIdx.x * blockDim.x + threadIdx.x;
    int edge = idx >> 5;
    int lane = idx & 31;
    if (edge < E) {
        int s = src[edge];
        int d = dst[edge];
        float nm = dinv[s] * dinv[d];
        const float4* hv = (const float4*)(h + (size_t)s * DHID) + lane;
        float4 v = *hv;
        float* o = out + (size_t)d * DHID + lane * 4;
        atomicAdd(o + 0, v.x * nm);
        atomicAdd(o + 1, v.y * nm);
        atomicAdd(o + 2, v.z * nm);
        atomicAdd(o + 3, v.w * nm);
    }
}

// ---------------- mean pool ----------------
__global__ void pool_acc(const float* __restrict__ f, const int* __restrict__ batch,
                         float* __restrict__ pooled, float* __restrict__ cnt, int N) {
    unsigned idx = blockIdx.x * blockDim.x + threadIdx.x;
    int node = idx >> 5;
    int lane = idx & 31;
    if (node < N) {
        int g = batch[node];
        const float4* fv = (const float4*)(f + (size_t)node * DHID) + lane;
        float4 v = *fv;
        float* o = pooled + (size_t)g * DHID + lane * 4;
        atomicAdd(o + 0, v.x);
        atomicAdd(o + 1, v.y);
        atomicAdd(o + 2, v.z);
        atomicAdd(o + 3, v.w);
        if (lane == 0) atomicAdd(cnt + g, 1.0f);
    }
}

__global__ void pool_div(const float* __restrict__ pooled, const float* __restrict__ cnt,
                         float* __restrict__ out) {
    int g = blockIdx.x;
    int c = threadIdx.x;
    float denom = fmaxf(cnt[g], 1.0f);
    out[g * DHID + c] = pooled[g * DHID + c] / denom;
}

// ---------------- classifier ----------------
__global__ void classifier_kernel(const float* __restrict__ f0, const float* __restrict__ f1,
                                  const float* __restrict__ Wc1, const float* __restrict__ bc1,
                                  const float* __restrict__ Wc2, const float* __restrict__ bc2,
                                  float* __restrict__ out) {
    __shared__ float hs[DHID];
    int g = blockIdx.x;
    int j = threadIdx.x;
    float acc = bc1[j];
    for (int k = 0; k < DHID; ++k) acc += f0[g * DHID + k] * Wc1[k * DHID + j];
    for (int k = 0; k < DHID; ++k) acc += f1[g * DHID + k] * Wc1[(DHID + k) * DHID + j];
    float hval = 1.0f / (1.0f + expf(-acc));
    hs[j] = hval * Wc2[j];
    __syncthreads();
    for (int off = DHID / 2; off > 0; off >>= 1) {
        if (j < off) hs[j] += hs[j + off];
        __syncthreads();
    }
    if (j == 0) out[g] = 1.0f / (1.0f + expf(-(hs[0] + bc2[0])));
}

// ---------------- host launch ----------------
extern "C" void kernel_launch(void* const* d_in, const int* in_sizes, int n_in,
                              void* d_out, int out_size, void* d_ws, size_t ws_size,
                              hipStream_t stream) {
    (void)n_in; (void)out_size; (void)ws_size;
    const float* x[2]     = { (const float*)d_in[0], (const float*)d_in[1] };
    const int*   ei[2]    = { (const int*)d_in[2],   (const int*)d_in[3] };
    const int*   batch[2] = { (const int*)d_in[4],   (const int*)d_in[5] };
    const float* W1  = (const float*)d_in[6];
    const float* b1  = (const float*)d_in[7];
    const float* W2  = (const float*)d_in[8];
    const float* b2  = (const float*)d_in[9];
    const float* Wc1 = (const float*)d_in[10];
    const float* bc1 = (const float*)d_in[11];
    const float* Wc2 = (const float*)d_in[12];
    const float* bc2 = (const float*)d_in[13];

    const int N = in_sizes[0] / DHID;   // DIN == DH == 128
    const int E = in_sizes[2] / 2;

    char* ws = (char*)d_ws;
    size_t off = 0;
    auto alloc = [&](size_t bytes) {
        size_t r = off;
        off += (bytes + 255) & ~(size_t)255;
        return r;
    };
    unsigned short* w1p   = (unsigned short*)(ws + alloc((size_t)DHID * DHID * 2));
    unsigned short* w2p   = (unsigned short*)(ws + alloc((size_t)DHID * DHID * 2));
    float* dinv   = (float*)(ws + alloc((size_t)N * 4));
    float* bufH   = (float*)(ws + alloc((size_t)N * DHID * 4));
    float* bufA   = (float*)(ws + alloc((size_t)N * DHID * 4));
    float* pooled = (float*)(ws + alloc((size_t)GSEG * DHID * 4));
    float* cnt    = (float*)(ws + alloc((size_t)GSEG * 4));
    float* femb   = (float*)(ws + alloc((size_t)2 * GSEG * DHID * 4));

    // Pack shared weights to bf16 fragment layout (once per call).
    pack_w_bf16<<<64, 256, 0, stream>>>(W1, w1p);
    pack_w_bf16<<<64, 256, 0, stream>>>(W2, w2p);

    const int total  = N * DHID;
    const int nBlkN  = (N + 255) / 256;
    const int nBlkE  = (E + 255) / 256;
    const int nBlkT  = (total + 255) / 256;
    const int nBlkE32 = (E * 32 + 255) / 256;
    const int nBlkN32 = (N * 32 + 255) / 256;

    for (int b = 0; b < 2; ++b) {
        const int* srcIdx = ei[b];
        const int* dstIdx = ei[b] + E;

        // degree with self-loop -> D^{-1/2}
        fill_f32<<<nBlkN, 256, 0, stream>>>(dinv, N, 1.0f);
        deg_scatter<<<nBlkE, 256, 0, stream>>>(dstIdx, dinv, E);
        rsqrt_inplace<<<nBlkN, 256, 0, stream>>>(dinv, N);

        // layer 1: H = X @ W1 ; A = norm-aggregate(H) + b1
        gemm_nk128_wmma<<<N / 16, 256, 0, stream>>>(x[b], (const U4*)w1p, bufH);
        self_bias<<<nBlkT, 256, 0, stream>>>(bufH, dinv, b1, bufA, total);
        agg_edges<<<nBlkE32, 256, 0, stream>>>(bufH, srcIdx, dstIdx, dinv, bufA, E);

        // layer 2: H = A @ W2 ; A = norm-aggregate(H) + b2
        gemm_nk128_wmma<<<N / 16, 256, 0, stream>>>(bufA, (const U4*)w2p, bufH);
        self_bias<<<nBlkT, 256, 0, stream>>>(bufH, dinv, b2, bufA, total);
        agg_edges<<<nBlkE32, 256, 0, stream>>>(bufH, srcIdx, dstIdx, dinv, bufA, E);

        // global mean pool
        fill_f32<<<(GSEG * DHID + 255) / 256, 256, 0, stream>>>(pooled, GSEG * DHID, 0.0f);
        fill_f32<<<1, GSEG, 0, stream>>>(cnt, GSEG, 0.0f);
        pool_acc<<<nBlkN32, 256, 0, stream>>>(bufA, batch[b], pooled, cnt, N);
        pool_div<<<GSEG, DHID, 0, stream>>>(pooled, cnt, femb + (size_t)b * GSEG * DHID);
    }

    classifier_kernel<<<GSEG, DHID, 0, stream>>>(
        femb, femb + (size_t)GSEG * DHID, Wc1, bc1, Wc2, bc2, (float*)d_out);
}
